// MiddleLayer_30545807409998
// MI455X (gfx1250) — matrix operand order repaired and checked by reading, not compiled, not run on hardware
//
#include <hip/hip_runtime.h>
#include <hip/hip_bf16.h>
#include <math.h>

typedef __bf16 bf16;
typedef __attribute__((ext_vector_type(16))) __bf16 v16bf;
typedef __attribute__((ext_vector_type(8)))  __bf16 v8bf;
typedef __attribute__((ext_vector_type(8)))  float  v8f;

#define BATCH  8
#define TSTEPS 32
#define MLEN   8
#define DMODEL 256
#define UNITS  2048   // MLEN*DMODEL
#define U3     6144   // 3*UNITS
#define NSEQ   64     // MLEN*BATCH
#define BT     256    // BATCH*TSTEPS
#define KSPLIT 4      // split-K factor for the recurrence GEMM (latency-bound)
#define KCHUNK (UNITS / KSPLIT)   // 512

__device__ __forceinline__ bf16 tobf(float f) { return (bf16)f; }

// ---- CDNA5 wave32 WMMA 16x16x32 bf16 fragment layouts ----
// A (16x32 row-major, lda elems): lane holds row r=lane&15.
//   elem j in [0,8):  K = kbase + (lane>>4)*8 + j          (contiguous -> one b128)
//   elem j in [8,16): K = kbase + 16 + (lane>>4)*8 + (j-8) (contiguous -> one b128)
// B is passed TRANSPOSED (Bt: row n holds its K-run contiguously, ldb elems):
//   lane holds col n=lane&15; same K mapping as A (two b128 loads)
__device__ __forceinline__ v8f wmma_kloop(const bf16* __restrict__ A, int lda,
                                          const bf16* __restrict__ Bt, int ldb,
                                          int K, v8f acc) {
  const int lane = threadIdx.x & 31;
  const int r    = lane & 15;
  const int hi   = lane >> 4;
  const bf16* arow = A  + (size_t)r * lda + hi * 8;
  const bf16* bcol = Bt + (size_t)r * ldb + hi * 8;
  for (int k = 0; k < K; k += 32) {
    v8bf alo = *(const v8bf*)(arow + k);
    v8bf ahi = *(const v8bf*)(arow + k + 16);
    v16bf a = __builtin_shufflevector(alo, ahi,
                0,1,2,3,4,5,6,7,8,9,10,11,12,13,14,15);
    v8bf blo = *(const v8bf*)(bcol + k);
    v8bf bhi = *(const v8bf*)(bcol + k + 16);
    v16bf b = __builtin_shufflevector(blo, bhi,
                0,1,2,3,4,5,6,7,8,9,10,11,12,13,14,15);
    acc = __builtin_amdgcn_wmma_f32_16x16x32_bf16(false, a, false, b,
                                                  (short)0, acc, false, false);
  }
  return acc;
}

// C/D 16x16 f32 tile: lane=col (lane&15), VGPR v -> row v + 8*(lane>>4)
__device__ __forceinline__ void store_tile(float* __restrict__ C, int ldc, v8f acc) {
  const int lane = threadIdx.x & 31;
  const int c  = lane & 15;
  const int hi = lane >> 4;
#pragma unroll
  for (int v = 0; v < 8; ++v) C[(size_t)(v + hi * 8) * ldc + c] = acc[v];
}

// f32 (K x N, row-major) -> bf16 transposed (N x K)
__global__ void k_cvt_transpose(const float* __restrict__ src, bf16* __restrict__ dst,
                                int K, int N) {
  int i = blockIdx.x * blockDim.x + threadIdx.x;
  if (i >= K * N) return;
  int k = i / N, n = i - k * N;
  dst[(size_t)n * K + k] = tobf(src[i]);
}

__global__ void k_cvt(const float* __restrict__ src, bf16* __restrict__ dst, int n) {
  int i = blockIdx.x * blockDim.x + threadIdx.x;
  if (i < n) dst[i] = tobf(src[i]);
}

// P[l][bt][u] = x[b,t,l,:] @ W[l*256:(l+1)*256, u]   (A=xb, B=Wt transposed)
__global__ void k_gemm_xproj(const bf16* __restrict__ xb, const bf16* __restrict__ Wt,
                             float* __restrict__ P) {
  int wave = threadIdx.x >> 5;
  int l    = blockIdx.z;
  int row0 = blockIdx.y * 16;
  int col0 = blockIdx.x * 64 + wave * 16;
  v8f acc = {};
  acc = wmma_kloop(xb + (size_t)row0 * UNITS + l * DMODEL, UNITS,
                   Wt + (size_t)col0 * UNITS + l * DMODEL, UNITS,
                   DMODEL, acc);
  store_tile(P + ((size_t)l * BT + row0) * U3 + col0, U3, acc);
}

// xp[(m*BATCH+b)][t][u] = b_in[u] + sum_{l<=m} P[l][b*T+t][u]
__global__ void k_prefix_xp(const float* __restrict__ P, const float* __restrict__ b_in,
                            float* __restrict__ xp) {
  int i = blockIdx.x * blockDim.x + threadIdx.x;
  if (i >= BT * U3) return;
  int bt = i / U3, u = i - bt * U3;
  int b = bt / TSTEPS, t = bt - b * TSTEPS;
  float acc = b_in[u];
#pragma unroll
  for (int m = 0; m < MLEN; ++m) {
    acc += P[((size_t)m * BT + bt) * U3 + u];
    xp[(((size_t)(m * BATCH + b)) * TSTEPS + t) * U3 + u] = acc;
  }
}

__global__ void k_init_h(const float* __restrict__ h0, float* __restrict__ H,
                         bf16* __restrict__ Hb) {
  int i = blockIdx.x * blockDim.x + threadIdx.x;
  if (i >= NSEQ * UNITS) return;
  int s = i / UNITS, k = i - s * UNITS;
  float v = h0[(size_t)(s & 7) * UNITS + k];
  H[i] = v;
  Hb[i] = tobf(v);
}

// Split-K recurrence GEMM: recp[z][s][u] = Hb[s][z*512 .. ) @ U[z*512.., u].
// 4x the waves (6144 vs 1536) and a 4x shorter WMMA dependency chain per wave —
// the recurrence is latency-bound (U is L2-resident), not bandwidth-bound.
__global__ void k_gemm_rec(const bf16* __restrict__ Hb, const bf16* __restrict__ Ut,
                           float* __restrict__ recp) {
  int wave = threadIdx.x >> 5;
  int kz   = blockIdx.z;
  int row0 = blockIdx.y * 16;
  int col0 = blockIdx.x * 64 + wave * 16;
  int k0   = kz * KCHUNK;
  v8f acc = {};
  acc = wmma_kloop(Hb + (size_t)row0 * UNITS + k0, UNITS,
                   Ut + (size_t)col0 * UNITS + k0, UNITS,
                   KCHUNK, acc);
  store_tile(recp + ((size_t)kz * NSEQ + row0) * U3 + col0, U3, acc);
}

// GRU gates (Keras v2, reset_after=True): reduce the KSPLIT partials, apply
// gates, and directly scatter the needed output slices
// (ret takes block m of sequence m; state takes m = MLEN-1 at t = T-1).
__global__ void k_gate(int t, const float* __restrict__ xp, const float* __restrict__ recp,
                       const float* __restrict__ b_rec,
                       float* __restrict__ H, bf16* __restrict__ Hb,
                       float* __restrict__ ret, float* __restrict__ state) {
  int i = blockIdx.x * blockDim.x + threadIdx.x;
  if (i >= NSEQ * UNITS) return;
  int s = i / UNITS, u = i - s * UNITS;
  int m = s >> 3, b = s & 7;
  const float* xps = xp + ((size_t)s * TSTEPS + t) * U3;
  float xz = xps[u];
  float xr = xps[UNITS + u];
  float xh = xps[2 * UNITS + u];
  float rz = b_rec[u];
  float rr = b_rec[UNITS + u];
  float rh = b_rec[2 * UNITS + u];
#pragma unroll
  for (int z = 0; z < KSPLIT; ++z) {
    const float* rs = recp + ((size_t)z * NSEQ + s) * U3;
    rz += rs[u];
    rr += rs[UNITS + u];
    rh += rs[2 * UNITS + u];
  }
  float zg = 1.0f / (1.0f + __expf(-(xz + rz)));
  float rg = 1.0f / (1.0f + __expf(-(xr + rr)));
  float hh = tanhf(xh + rg * rh);
  float h  = H[i];
  float hn = zg * h + (1.0f - zg) * hh;
  H[i]  = hn;
  Hb[i] = tobf(hn);
  if ((u >> 8) == m)
    ret[(((size_t)b * TSTEPS + t) * MLEN + m) * DMODEL + (u & 255)] = hn;
  if (t == TSTEPS - 1 && m == MLEN - 1)
    state[(size_t)b * UNITS + u] = hn;
}

extern "C" void kernel_launch(void* const* d_in, const int* in_sizes, int n_in,
                              void* d_out, int out_size, void* d_ws, size_t ws_size,
                              hipStream_t stream) {
  const float* x  = (const float*)d_in[0];  // (8,32,8,256)
  const float* h0 = (const float*)d_in[1];  // (8,2048)
  const float* W  = (const float*)d_in[2];  // (2048,6144)
  const float* Uu = (const float*)d_in[3];  // (2048,6144)
  const float* bb = (const float*)d_in[4];  // (2,6144)

  float* ret   = (float*)d_out;                                   // (8,32,8,256)
  float* state = ret + (size_t)BATCH * TSTEPS * MLEN * DMODEL;    // (8,2048)

  char* ws = (char*)d_ws;
  size_t off = 0;
  auto alloc = [&](size_t bytes) -> void* {
    off = (off + 255) & ~(size_t)255;
    void* p = ws + off;
    off += bytes;
    return p;
  };
  bf16*  Wt   = (bf16*) alloc((size_t)U3 * UNITS * sizeof(bf16));   // W^T bf16
  bf16*  Ut   = (bf16*) alloc((size_t)U3 * UNITS * sizeof(bf16));   // U^T bf16 (L2-resident)
  bf16*  xb   = (bf16*) alloc((size_t)BT * UNITS * sizeof(bf16));   // x bf16
  float* P    = (float*)alloc((size_t)MLEN * BT * U3 * sizeof(float));
  float* xp   = (float*)alloc((size_t)NSEQ * TSTEPS * U3 * sizeof(float));
  float* H    = (float*)alloc((size_t)NSEQ * UNITS * sizeof(float));
  bf16*  Hb   = (bf16*) alloc((size_t)NSEQ * UNITS * sizeof(bf16));
  float* recp = (float*)alloc((size_t)KSPLIT * NSEQ * U3 * sizeof(float));

  const int nWU = UNITS * U3;
  k_cvt_transpose<<<(nWU + 255) / 256, 256, 0, stream>>>(W,  Wt, UNITS, U3);
  k_cvt_transpose<<<(nWU + 255) / 256, 256, 0, stream>>>(Uu, Ut, UNITS, U3);
  k_cvt<<<(BT * UNITS + 255) / 256, 256, 0, stream>>>(x, xb, BT * UNITS);

  // 8 small GEMMs instead of one masked 2048x2048x6144 GEMM (8x fewer FLOPs)
  k_gemm_xproj<<<dim3(U3 / 64, BT / 16, MLEN), 128, 0, stream>>>(xb, Wt, P);
  k_prefix_xp<<<(BT * U3 + 255) / 256, 256, 0, stream>>>(P, bb /*b_in*/, xp);
  k_init_h<<<(NSEQ * UNITS + 255) / 256, 256, 0, stream>>>(h0, H, Hb);

  // Sequential GRU recurrence: 32 dependent (64x2048)@(2048x6144) bf16-WMMA
  // GEMMs, split-K by 4 to shorten the latency-critical chain.
  for (int t = 0; t < TSTEPS; ++t) {
    k_gemm_rec<<<dim3(U3 / 64, NSEQ / 16, KSPLIT), 128, 0, stream>>>(Hb, Ut, recp);
    k_gate<<<(NSEQ * UNITS + 255) / 256, 256, 0, stream>>>(t, xp, recp, bb + U3 /*b_rec*/,
                                                           H, Hb, ret, state);
  }
}
